// MiniTFT_Step8_77412490543362
// MI455X (gfx1250) — matrix operand order repaired and checked by reading, not compiled, not run on hardware
//
#include <hip/hip_runtime.h>

// ---------------- CDNA5 WMMA types / helpers ----------------
typedef __attribute__((ext_vector_type(16))) __bf16 v16bf;
typedef __attribute__((ext_vector_type(8)))  float  v8f;
typedef unsigned int u32;
typedef __attribute__((ext_vector_type(4))) u32 u32x4;
typedef __attribute__((ext_vector_type(8))) u32 u32x8;

struct __align__(16) U4 { unsigned int x, y, z, w; };

#define WMMA_BF16(A, B, C) \
  __builtin_amdgcn_wmma_f32_16x16x32_bf16(false, (A), false, (B), (short)0, (C), false, false)

// butterfly exchange within 16-lane groups: ds_swizzle group-of-32, and=0x1F, xor=s
#define SWZ_F(x, imm) __int_as_float(__builtin_amdgcn_ds_swizzle(__float_as_int((x)), (imm)))

#if __has_builtin(__builtin_amdgcn_s_wait_tensorcnt)
#define WAIT_TENSOR0() __builtin_amdgcn_s_wait_tensorcnt(0)
#else
#define WAIT_TENSOR0() asm volatile("s_wait_tensorcnt 0x0" ::: "memory")
#endif

union FragBits { v16bf v; struct { U4 lo; U4 hi; } b; };

// A fragment (16x32 bf16, row-major src, row stride ld).
// ISA 7.12.2: lane l: M=l&15, K = kb..kb+7 (elems0-7) and kb+16..kb+23 (elems8-15), kb = koff+8*(l>=16)
__device__ __forceinline__ v16bf load_frag_a(const __bf16* src, int ld, int koff) {
  const int l = threadIdx.x & 31;
  const __bf16* p = src + (l & 15) * ld + koff + ((l >> 4) << 3);
  FragBits f;
  f.b.lo = *(const U4*)p;
  f.b.hi = *(const U4*)(p + 16);
  return f.v;
}

// B fragment (32x16 bf16 KxN) from an (N,K)-major source (weight row n holds K values).
// lane l: N=l&15, K = kb..kb+7 (elems0-7) and kb+16..kb+23 (elems8-15), kb = koff+8*(l>>4).
__device__ __forceinline__ v16bf load_frag_b(const __bf16* srcNK, int ld, int koff) {
  const int l = threadIdx.x & 31;
  const __bf16* p = srcNK + (l & 15) * ld + koff + ((l >> 4) << 3);
  FragBits f;
  f.b.lo = *(const U4*)p;
  f.b.hi = *(const U4*)(p + 16);
  return f.v;
}

__device__ __forceinline__ float sigf(float x) { return 1.0f / (1.0f + __expf(-x)); }

// ---------------- Tensor Data Mover: 2D tile load (256 cols x 16 batch, bf16) ----------------
// Source tensor: 2D [256][128] bf16 (dim0 = batch, contiguous, stride 128).
// Tile: tile_dim0 = 16 (batch), tile_dim1 = 256 (gate cols) -> packed 16-elem rows in LDS.
__device__ __forceinline__ void tdm_load_gates(u32 lds_byte_off, const void* gaddr) {
  const unsigned long long ga = (unsigned long long)(uintptr_t)gaddr;
  u32x4 g0;
  g0[0] = 1u;                                            // count=1, user mode, no gather
  g0[1] = lds_byte_off;                                  // lds_addr
  g0[2] = (u32)(ga & 0xFFFFFFFFu);                       // global_addr[31:0]
  g0[3] = (u32)((ga >> 32) & 0x01FFFFFFu) | (2u << 30);  // global_addr[56:32] | type=2
  u32x8 g1;
  g1[0] = (1u << 16);     // workgroup_mask=0 | data_size=1 (2 bytes)
  g1[1] = (128u << 16);   // tensor_dim0[15:0]=128 (bits 63:48)
  g1[2] = (256u << 16);   // tensor_dim0 hi=0 | tensor_dim1[15:0]=256
  g1[3] = (16u << 16);    // tensor_dim1 hi=0 | tile_dim0=16
  g1[4] = 256u;           // tile_dim1=256 | tile_dim2=0
  g1[5] = 128u;           // tensor_dim0_stride[31:0]=128
  g1[6] = 0u;             // stride0 hi | tensor_dim1_stride lo (unused, 2D tile)
  g1[7] = 0u;
  asm volatile("tensor_load_to_lds %0, %1" :: "s"(g0), "s"(g1) : "memory");
}

// ---------------- problem dims ----------------
// B=128, TE=512, TD=128, F=32, H=64, O=1
// rows are t-major: row = t*128 + b   (16-row tiles never straddle t since 16|128)

// ---------------- workspace layout (bytes) ----------------
constexpr size_t OFF_ENCH   = 0;                                      // bf16 [65536][64]
constexpr size_t OFF_DECH   = OFF_ENCH   + (size_t)65536 * 64 * 2;    // bf16 [16384][64]
constexpr size_t OFF_DECHF  = OFF_DECH   + (size_t)16384 * 64 * 2;    // f32  [16384][64]
constexpr size_t OFF_GENC   = OFF_DECHF  + (size_t)16384 * 64 * 4;    // bf16 [512][256][128]
constexpr size_t OFF_GDEC   = OFF_GENC   + (size_t)512 * 256 * 128 * 2; // bf16 [128][256][128]
constexpr size_t OFF_DECOUT = OFF_GDEC   + (size_t)128 * 256 * 128 * 2; // bf16 [16384][64]
constexpr size_t OFF_WIHE   = OFF_DECOUT + (size_t)16384 * 64 * 2;    // bf16 [256][64]
constexpr size_t OFF_WHHE   = OFF_WIHE + 16384 * 2;
constexpr size_t OFF_WIHD   = OFF_WHHE + 16384 * 2;
constexpr size_t OFF_WHHD   = OFF_WIHD + 16384 * 2;
constexpr size_t OFF_WPRET  = OFF_WHHD + 16384 * 2;                   // bf16 [64][32] (transposed W_pre)
constexpr size_t OFF_BPRET  = OFF_WPRET + 2048 * 2;                   // bf16 [64][32]
constexpr size_t OFF_SE     = OFF_BPRET + 2048 * 2;                   // bf16 [32][32]
constexpr size_t OFF_SD     = OFF_SE + 1024 * 2;
constexpr size_t OFF_WZ     = OFF_SD + 1024 * 2;                      // bf16 [64][64]
constexpr size_t OFF_WG     = OFF_WZ + 4096 * 2;
constexpr size_t OFF_GBE    = OFF_WG + 4096 * 2;                      // f32 [256] combined bih+bhh (enc)
constexpr size_t OFF_GBD    = OFF_GBE + 256 * 4;                      // f32 [256] (dec)

// ---------------- K0: weight conversion + trivial h0/c0 outputs ----------------
__global__ void __launch_bounds__(256) prep_kernel(
    char* ws,
    const float* Wih_e, const float* Whh_e, const float* bih_e, const float* bhh_e,
    const float* Wih_d, const float* Whh_d, const float* bih_d, const float* bhh_d,
    const float* W_pre, const float* b_pre, const float* S_enc, const float* S_dec,
    const float* Wz, const float* Wg, const float* b_h0, const float* b_c0,
    float* out) {
  __bf16* WihE  = (__bf16*)(ws + OFF_WIHE);
  __bf16* WhhE  = (__bf16*)(ws + OFF_WHHE);
  __bf16* WihD  = (__bf16*)(ws + OFF_WIHD);
  __bf16* WhhD  = (__bf16*)(ws + OFF_WHHD);
  __bf16* WpreT = (__bf16*)(ws + OFF_WPRET);
  __bf16* bpreT = (__bf16*)(ws + OFF_BPRET);
  __bf16* Sebf  = (__bf16*)(ws + OFF_SE);
  __bf16* Sdbf  = (__bf16*)(ws + OFF_SD);
  __bf16* Wzbf  = (__bf16*)(ws + OFF_WZ);
  __bf16* Wgbf  = (__bf16*)(ws + OFF_WG);
  float*  gbe   = (float*)(ws + OFF_GBE);
  float*  gbd   = (float*)(ws + OFF_GBD);

  const int gid = blockIdx.x * blockDim.x + threadIdx.x;
  const int stride = gridDim.x * blockDim.x;
  for (int i = gid; i < 16384; i += stride) {
    WihE[i] = (__bf16)Wih_e[i]; WhhE[i] = (__bf16)Whh_e[i];
    WihD[i] = (__bf16)Wih_d[i]; WhhD[i] = (__bf16)Whh_d[i];
  }
  for (int i = gid; i < 2048; i += stride) {   // (F=32,H=64) -> (H,F)
    int f = i >> 6, h = i & 63;
    WpreT[h * 32 + f] = (__bf16)W_pre[i];
    bpreT[h * 32 + f] = (__bf16)b_pre[i];
  }
  for (int i = gid; i < 1024; i += stride) { Sebf[i] = (__bf16)S_enc[i]; Sdbf[i] = (__bf16)S_dec[i]; }
  for (int i = gid; i < 4096; i += stride) { Wzbf[i] = (__bf16)Wz[i]; Wgbf[i] = (__bf16)Wg[i]; }
  for (int i = gid; i < 256; i += stride) { gbe[i] = bih_e[i] + bhh_e[i]; gbd[i] = bih_d[i] + bhh_d[i]; }
  // outputs h0 = b_h0 (B,H), c0 = b_c0 (static input is zeros)
  for (int i = gid; i < 8192; i += stride) {
    int h = i & 63;
    out[16384 + i] = b_h0[h];
    out[24576 + i] = b_c0[h];
  }
}

// ---------------- K1: VSN (softmax feature selection + projection), WMMA ----------------
// one wave per 16-row tile; 8 waves per block
__global__ void __launch_bounds__(256) vsn_kernel(char* ws, const float* encX, const float* decX) {
  const __bf16* Sebf  = (const __bf16*)(ws + OFF_SE);
  const __bf16* Sdbf  = (const __bf16*)(ws + OFF_SD);
  const __bf16* WpreT = (const __bf16*)(ws + OFF_WPRET);
  const __bf16* bpreT = (const __bf16*)(ws + OFF_BPRET);
  __bf16* encH  = (__bf16*)(ws + OFF_ENCH);
  __bf16* decH  = (__bf16*)(ws + OFF_DECH);
  float*  decHf = (float*)(ws + OFF_DECHF);

  __shared__ float  xs[8][16][32];
  __shared__ __bf16 xb[8][16][32];
  __shared__ __bf16 wsm[8][16][32];
  __shared__ __bf16 xw[8][16][32];

  const int wv = threadIdx.x >> 5, lane = threadIdx.x & 31;
  const int tile = blockIdx.x * 8 + wv;                 // 0..5119
  const bool isEnc = tile < 4096;
  const int  lt = isEnc ? tile : tile - 4096;
  const int  T  = isEnc ? 512 : 128;
  const float*  X   = isEnc ? encX : decX;
  const __bf16* Sbf = isEnc ? Sebf : Sdbf;
  const int row0 = lt * 16, t = row0 >> 7, bb = row0 & 127;

  { // stage x tile into LDS (f32 for exact elementwise, bf16 for WMMA A)
    const int m = lane & 15, half = lane >> 4;
    const float* xp = X + ((size_t)(bb + m) * T + t) * 32 + half * 16;
#pragma unroll
    for (int i = 0; i < 16; ++i) {
      float v = xp[i];
      xs[wv][m][half * 16 + i] = v;
      xb[wv][m][half * 16 + i] = (__bf16)v;
    }
  }
  __syncthreads();

  const v16bf ax = load_frag_a(&xb[wv][0][0], 32, 0);
  v8f zero = {};
  v8f L0 = WMMA_BF16(ax, load_frag_b(Sbf, 32, 0), zero);            // logits cols 0..15
  v8f L1 = WMMA_BF16(ax, load_frag_b(Sbf + 16 * 32, 32, 0), zero);  // logits cols 16..31

  const int n = lane & 15, mh = (lane >> 4) << 3;
#pragma unroll
  for (int r = 0; r < 8; ++r) {
    float a0 = L0[r], a1 = L1[r];
    float mx = fmaxf(a0, a1);
    mx = fmaxf(mx, SWZ_F(mx, 0x201F));   // xor 8
    mx = fmaxf(mx, SWZ_F(mx, 0x101F));   // xor 4
    mx = fmaxf(mx, SWZ_F(mx, 0x081F));   // xor 2
    mx = fmaxf(mx, SWZ_F(mx, 0x041F));   // xor 1
    float e0 = __expf(a0 - mx), e1 = __expf(a1 - mx);
    float sm = e0 + e1;
    sm += SWZ_F(sm, 0x201F);
    sm += SWZ_F(sm, 0x101F);
    sm += SWZ_F(sm, 0x081F);
    sm += SWZ_F(sm, 0x041F);
    float inv = 1.0f / sm, w0 = e0 * inv, w1 = e1 * inv;
    int m = r + mh;
    wsm[wv][m][n]      = (__bf16)w0;
    wsm[wv][m][n + 16] = (__bf16)w1;
    xw[wv][m][n]       = (__bf16)(w0 * xs[wv][m][n]);
    xw[wv][m][n + 16]  = (__bf16)(w1 * xs[wv][m][n + 16]);
  }
  __syncthreads();

  const v16bf aw  = load_frag_a(&wsm[wv][0][0], 32, 0);
  const v16bf axw = load_frag_a(&xw[wv][0][0], 32, 0);
#pragma unroll
  for (int hb = 0; hb < 4; ++hb) {
    v8f acc = {};
    acc = WMMA_BF16(aw,  load_frag_b(bpreT + hb * 16 * 32, 32, 0), acc); // w @ b_pre
    acc = WMMA_BF16(axw, load_frag_b(WpreT + hb * 16 * 32, 32, 0), acc); // (x*w) @ W_pre
#pragma unroll
    for (int r = 0; r < 8; ++r) {
      int m = r + mh;
      size_t idx = (size_t)(row0 + m) * 64 + hb * 16 + n;
      if (isEnc) {
        encH[idx] = (__bf16)acc[r];
      } else {
        decH[idx]  = (__bf16)acc[r];
        decHf[idx] = acc[r];
      }
    }
  }
}

// ---------------- K2: precompute G = x @ Wih^T + (bih+bhh) for all timesteps ----------------
// stored as G[t][gate_col 256][batch 128] so LSTM consumes one contiguous 2D tile per step
__global__ void __launch_bounds__(256) xgemm_kernel(char* ws) {
  const __bf16* encH = (const __bf16*)(ws + OFF_ENCH);
  const __bf16* decH = (const __bf16*)(ws + OFF_DECH);
  const __bf16* WihE = (const __bf16*)(ws + OFF_WIHE);
  const __bf16* WihD = (const __bf16*)(ws + OFF_WIHD);
  const float*  gbe  = (const float*)(ws + OFF_GBE);
  const float*  gbd  = (const float*)(ws + OFF_GBD);
  __bf16* Genc = (__bf16*)(ws + OFF_GENC);
  __bf16* Gdec = (__bf16*)(ws + OFF_GDEC);

  const int wv = threadIdx.x >> 5, lane = threadIdx.x & 31;
  const int wid = blockIdx.x * 8 + wv;            // 0..81919
  const bool isEnc = wid < 65536;
  const int local = isEnc ? wid : wid - 65536;
  const int mt = local >> 4, nt = local & 15;

  const __bf16* A = (isEnc ? encH : decH) + (size_t)mt * 16 * 64;
  const __bf16* W = (isEnc ? WihE : WihD) + nt * 16 * 64;
  const float*  gb = isEnc ? gbe : gbd;
  __bf16* G = isEnc ? Genc : Gdec;

  const int row0 = mt * 16, t = row0 >> 7, bb = row0 & 127;
  const int n = lane & 15, half = lane >> 4;

  const v16bf a0 = load_frag_a(A, 64, 0);
  const v16bf a1 = load_frag_a(A, 64, 32);
  const v16bf b0 = load_frag_b(W, 64, 0);
  const v16bf b1 = load_frag_b(W, 64, 32);

  const float bias = gb[nt * 16 + n];
  v8f acc;
#pragma unroll
  for (int r = 0; r < 8; ++r) acc[r] = bias;
  acc = WMMA_BF16(a0, b0, acc);
  acc = WMMA_BF16(a1, b1, acc);

  union { __bf16 h[8]; U4 u; } st;
#pragma unroll
  for (int r = 0; r < 8; ++r) st.h[r] = (__bf16)acc[r];
  *(U4*)(G + ((size_t)(t * 256 + nt * 16 + n)) * 128 + bb + (half << 3)) = st.u;
}

// ---------------- K3: recurrent LSTM (encoder then decoder), WMMA per step ----------------
// 8 blocks x 128 threads; block owns 16 batch rows; wave w owns h-block w and all 4 gates.
// Gate preactivations stream in via double-buffered TDM (tensor_load_to_lds + s_wait_tensorcnt).
__global__ void __launch_bounds__(128) lstm_kernel(char* ws, const float* b_h0, const float* b_c0) {
  const __bf16* Genc = (const __bf16*)(ws + OFF_GENC);
  const __bf16* Gdec = (const __bf16*)(ws + OFF_GDEC);
  const __bf16* WhhE = (const __bf16*)(ws + OFF_WHHE);
  const __bf16* WhhD = (const __bf16*)(ws + OFF_WHHD);
  __bf16* decOut = (__bf16*)(ws + OFF_DECOUT);

  __shared__ __bf16 hs[16][64];        // shared hidden state (batch-tile x H)
  __shared__ __bf16 gtile[2][4096];    // double-buffered TDM tiles: [col 256][batch 16]

  const int wv = threadIdx.x >> 5, lane = threadIdx.x & 31;
  const int hb = wv;                          // h-block 0..3
  const int bb = blockIdx.x * 16;             // batch-tile base
  const int n = lane & 15, half = lane >> 4;

  for (int i = threadIdx.x; i < 1024; i += 128) hs[i >> 6][i & 63] = (__bf16)b_h0[i & 63];

  v8f c;
#pragma unroll
  for (int r = 0; r < 8; ++r) c[r] = b_c0[hb * 16 + n];

  v16bf Bf[4][2];
#pragma unroll
  for (int g = 0; g < 4; ++g) {
    Bf[g][0] = load_frag_b(WhhE + (g * 64 + hb * 16) * 64, 64, 0);
    Bf[g][1] = load_frag_b(WhhE + (g * 64 + hb * 16) * 64, 64, 32);
  }
  __syncthreads();

  const u32 lds0 = (u32)(uintptr_t)(const void*)&gtile[0][0];
  const u32 lds1 = (u32)(uintptr_t)(const void*)&gtile[1][0];

  for (int phase = 0; phase < 2; ++phase) {
    const __bf16* G = phase ? Gdec : Genc;
    const int Tn = phase ? 128 : 512;
    if (phase) {
#pragma unroll
      for (int g = 0; g < 4; ++g) {
        Bf[g][0] = load_frag_b(WhhD + (g * 64 + hb * 16) * 64, 64, 0);
        Bf[g][1] = load_frag_b(WhhD + (g * 64 + hb * 16) * 64, 64, 32);
      }
    }
    tdm_load_gates(lds0, G + bb);   // prologue: tile for t=0

    for (int t = 0; t < Tn; ++t) {
      const int cur = t & 1;
      WAIT_TENSOR0();               // own TDM for step t complete
      __syncthreads();              // also publishes previous step's hs
      if (t + 1 < Tn)
        tdm_load_gates(cur ? lds0 : lds1, G + (size_t)(t + 1) * 256 * 128 + bb);

      v8f acc[4];
#pragma unroll
      for (int g = 0; g < 4; ++g) {
        const __bf16* lp = &gtile[cur][(u32)(g * 64 + hb * 16 + n) * 16 + (half << 3)];
        union { U4 u; __bf16 h[8]; } ld;
        ld.u = *(const U4*)lp;
#pragma unroll
        for (int r = 0; r < 8; ++r) acc[g][r] = (float)ld.h[r];
      }

      const v16bf a0 = load_frag_a(&hs[0][0], 64, 0);
      const v16bf a1 = load_frag_a(&hs[0][0], 64, 32);
#pragma unroll
      for (int g = 0; g < 4; ++g) {
        acc[g] = WMMA_BF16(a0, Bf[g][0], acc[g]);
        acc[g] = WMMA_BF16(a1, Bf[g][1], acc[g]);
      }
      __syncthreads();   // everyone done reading hs (and gtile[cur])
#pragma unroll
      for (int r = 0; r < 8; ++r) {
        float iv = sigf(acc[0][r]);
        float fv = sigf(acc[1][r]);
        float gv = tanhf(acc[2][r]);
        float ov = sigf(acc[3][r]);
        float cv = fv * c[r] + iv * gv;
        c[r] = cv;
        float hv = ov * tanhf(cv);
        int m = r + (half << 3);
        __bf16 hvb = (__bf16)hv;
        hs[m][hb * 16 + n] = hvb;
        if (phase) decOut[((size_t)(t * 128 + bb + m)) * 64 + hb * 16 + n] = hvb;
      }
    }
  }
}

// ---------------- K4: gating + residual + layernorm + head ----------------
__global__ void __launch_bounds__(256) post_kernel(
    char* ws, const float* bz, const float* bg, const float* gamma, const float* beta,
    const float* W_head, const float* b_head, float* out) {
  const __bf16* decOut = (const __bf16*)(ws + OFF_DECOUT);
  const float*  decHf  = (const float*)(ws + OFF_DECHF);
  const __bf16* Wzbf   = (const __bf16*)(ws + OFF_WZ);
  const __bf16* Wgbf   = (const __bf16*)(ws + OFF_WG);

  __shared__ float ys[8][16][64];

  const int wv = threadIdx.x >> 5, lane = threadIdx.x & 31;
  const int tile = blockIdx.x * 8 + wv;   // 0..1023
  const int row0 = tile * 16;
  const int n = lane & 15, mh = (lane >> 4) << 3;

  const __bf16* A = decOut + (size_t)row0 * 64;
  const v16bf a0 = load_frag_a(A, 64, 0);
  const v16bf a1 = load_frag_a(A, 64, 32);

#pragma unroll
  for (int hbk = 0; hbk < 4; ++hbk) {
    v8f zacc, gacc;
    const float zb = bz[hbk * 16 + n], gb2 = bg[hbk * 16 + n];
#pragma unroll
    for (int r = 0; r < 8; ++r) { zacc[r] = zb; gacc[r] = gb2; }
    zacc = WMMA_BF16(a0, load_frag_b(Wzbf + hbk * 16 * 64, 64, 0), zacc);
    zacc = WMMA_BF16(a1, load_frag_b(Wzbf + hbk * 16 * 64, 64, 32), zacc);
    gacc = WMMA_BF16(a0, load_frag_b(Wgbf + hbk * 16 * 64, 64, 0), gacc);
    gacc = WMMA_BF16(a1, load_frag_b(Wgbf + hbk * 16 * 64, 64, 32), gacc);
#pragma unroll
    for (int r = 0; r < 8; ++r) {
      int m = r + mh, h = hbk * 16 + n;
      float u = sigf(gacc[r]) * zacc[r];
      ys[wv][m][h] = u + decHf[(size_t)(row0 + m) * 64 + h];
    }
  }
  __syncthreads();

  if (lane < 16) {
    const int m = lane, row = row0 + m;
    float mu = 0.f;
#pragma unroll
    for (int h = 0; h < 64; ++h) mu += ys[wv][m][h];
    mu *= (1.0f / 64.0f);
    float var = 0.f;
#pragma unroll
    for (int h = 0; h < 64; ++h) { float d = ys[wv][m][h] - mu; var += d * d; }
    var *= (1.0f / 64.0f);
    const float inv = rsqrtf(var + 1e-5f);
    float accp = b_head[0];
#pragma unroll
    for (int h = 0; h < 64; ++h) {
      float y = (ys[wv][m][h] - mu) * inv * gamma[h] + beta[h];
      accp += y * W_head[h];
    }
    const int b = row & 127, tt = row >> 7;   // rows are t-major; output is (B,TD,1)
    out[b * 128 + tt] = accp;
  }
}

// ---------------- launcher ----------------
extern "C" void kernel_launch(void* const* d_in, const int* in_sizes, int n_in,
                              void* d_out, int out_size, void* d_ws, size_t ws_size,
                              hipStream_t stream) {
  (void)in_sizes; (void)n_in; (void)out_size; (void)ws_size;
  const float* encX  = (const float*)d_in[0];
  const float* decX  = (const float*)d_in[1];
  const float* W_pre = (const float*)d_in[2];
  const float* b_pre = (const float*)d_in[3];
  const float* S_enc = (const float*)d_in[4];
  const float* S_dec = (const float*)d_in[5];
  const float* b_h0  = (const float*)d_in[7];
  const float* b_c0  = (const float*)d_in[9];
  const float* Wih_e = (const float*)d_in[10];
  const float* Whh_e = (const float*)d_in[11];
  const float* bih_e = (const float*)d_in[12];
  const float* bhh_e = (const float*)d_in[13];
  const float* Wih_d = (const float*)d_in[14];
  const float* Whh_d = (const float*)d_in[15];
  const float* bih_d = (const float*)d_in[16];
  const float* bhh_d = (const float*)d_in[17];
  const float* Wz    = (const float*)d_in[18];
  const float* bz    = (const float*)d_in[19];
  const float* Wg    = (const float*)d_in[20];
  const float* bg    = (const float*)d_in[21];
  const float* gamma = (const float*)d_in[22];
  const float* beta  = (const float*)d_in[23];
  const float* Whead = (const float*)d_in[24];
  const float* bhead = (const float*)d_in[25];
  float* out = (float*)d_out;
  char*  ws  = (char*)d_ws;

  prep_kernel<<<64, 256, 0, stream>>>(ws, Wih_e, Whh_e, bih_e, bhh_e,
                                      Wih_d, Whh_d, bih_d, bhh_d,
                                      W_pre, b_pre, S_enc, S_dec, Wz, Wg, b_h0, b_c0, out);
  vsn_kernel<<<640, 256, 0, stream>>>(ws, encX, decX);          // 5120 tiles, 1 wave each
  xgemm_kernel<<<10240, 256, 0, stream>>>(ws);                  // 81920 output tiles
  lstm_kernel<<<8, 128, 0, stream>>>(ws, b_h0, b_c0);           // 8 batch-tiles, 4 waves each
  post_kernel<<<128, 256, 0, stream>>>(ws, bz, bg, gamma, beta, Whead, bhead, out);
}
